// MutualMaskGenerator_55679956025685
// MI455X (gfx1250) — compile-verified, matrix operand order checked by reading, not run on hardware
//
#include <hip/hip_runtime.h>
#include <hip/hip_bf16.h>

// ---------------------------------------------------------------------------
// MutualMaskGenerator on MI455X (gfx1250, wave32, WMMA).
// All large matmuls go through v_wmma_f32_16x16x32_f16 (f16 in, f32 accum),
// with double-buffered LDS tiles so global loads overlap WMMA issue.
// ---------------------------------------------------------------------------

typedef __attribute__((ext_vector_type(16))) _Float16 v16h;
typedef __attribute__((ext_vector_type(8)))  float    v8f;

#define Bsz   32
#define Ssz   256
#define Esz   768
#define Hn    12
#define DHn   64
#define Lnum  2
#define FFn   2048
#define ROWS  (Bsz * Ssz)          // 8192 tokens
#define NEGV  (-1e30f)
#define KEEPH 64
#define KEEPT 63

#define BKt 32                     // K step (one 16x16x32 WMMA deep)

// --------------------------- reductions ------------------------------------
__device__ __forceinline__ float block_sum(float v, float* red) {
  int i = threadIdx.x;
  red[i] = v; __syncthreads();
  for (int s = 128; s > 0; s >>= 1) { if (i < s) red[i] += red[i + s]; __syncthreads(); }
  float r = red[0]; __syncthreads();
  return r;
}
__device__ __forceinline__ float block_max(float v, float* red) {
  int i = threadIdx.x;
  red[i] = v; __syncthreads();
  for (int s = 128; s > 0; s >>= 1) { if (i < s) red[i] = fmaxf(red[i], red[i + s]); __syncthreads(); }
  float r = red[0]; __syncthreads();
  return r;
}

// --------------------------- WMMA GEMM core --------------------------------
// BT=1:  C[M,N] = alpha * A[M,K] @ W[N,K]^T   (+bias, act)
// BT=0:  C[M,N] = alpha * A[M,K] @ Bm[K,N]    (+bias, act)
// 256 threads = 8 waves arranged WM x WN; each wave owns (MI*16) x (NJ*16).
// Ping-pong LDS: next tile's global loads are issued before this tile's
// WMMAs and retired into the alternate buffer after them -> one barrier/step.
template<int BT, int WM, int WN, int MI, int NJ>
__device__ __forceinline__ void gemm_core(
    const _Float16* __restrict__ A, const _Float16* __restrict__ Bm,
    const float* __restrict__ bias, float* __restrict__ Cf, _Float16* __restrict__ Ch,
    int K, int lda, int ldb, int ldc, float alpha, int act, int tm0, int tn0)
{
  constexpr int BM = WM * MI * 16;
  constexpr int BN = WN * NJ * 16;
  static_assert(WM * WN == 8, "8 waves per block");
  static_assert(BM == 128, "A-tile loader assumes 128 rows");
  static_assert(BT ? (BN == 128) : (BN == 64), "B-tile loader shape");
  constexpr int SB_ROWS = BT ? BN : BKt;
  constexpr int SB_STR  = BT ? 40 : (BN + 8);   // padded strides (bank-conflict free)
  constexpr int SA_SZ   = BM * 40;
  constexpr int SB_SZ   = SB_ROWS * SB_STR;

  __shared__ _Float16 sA[2 * SA_SZ];
  __shared__ _Float16 sB[2 * SB_SZ];

  const int tid  = threadIdx.x;
  const int wave = tid >> 5;
  const int lane = tid & 31;
  const int wm   = wave % WM;
  const int wn   = wave / WM;
  const int half = lane >> 4;       // hi/lo 16-lane group
  const int lx   = lane & 15;

  const int ar = tid >> 1, ap = tid & 1;   // 128-row x 32-half cooperative map
  const int br = tid >> 3, bp = tid & 7;   // 32-row x 64-half cooperative map

  uint4 ra0, ra1, rb0, rb1;

  auto gload = [&](int kb) {               // global -> registers (b128 loads)
    const uint4* ga = reinterpret_cast<const uint4*>(
        A + (size_t)(tm0 + ar) * lda + kb + ap * 16);
    ra0 = ga[0]; ra1 = ga[1];
    if (BT) {
      const uint4* gb = reinterpret_cast<const uint4*>(
          Bm + (size_t)(tn0 + ar) * ldb + kb + ap * 16);
      rb0 = gb[0]; rb1 = gb[1];
    } else {
      const uint4* gb = reinterpret_cast<const uint4*>(
          Bm + (size_t)(kb + br) * ldb + tn0 + bp * 8);
      rb0 = gb[0];
    }
  };
  auto sstore = [&](int buf) {             // registers -> LDS (b128 stores)
    uint4* s = reinterpret_cast<uint4*>(&sA[buf * SA_SZ + ar * 40 + ap * 16]);
    s[0] = ra0; s[1] = ra1;
    if (BT) {
      uint4* t = reinterpret_cast<uint4*>(&sB[buf * SB_SZ + ar * 40 + ap * 16]);
      t[0] = rb0; t[1] = rb1;
    } else {
      *reinterpret_cast<uint4*>(&sB[buf * SB_SZ + br * SB_STR + bp * 8]) = rb0;
    }
  };

  v8f acc[MI][NJ] = {};

  gload(0);
  sstore(0);
  __syncthreads();

  const int nk = K / BKt;
  for (int t = 0; t < nk; ++t) {
    const int cur = t & 1, nxt = cur ^ 1;
    const bool more = (t + 1) < nk;
    if (more) gload((t + 1) * BKt);        // overlap HBM/L2 latency with WMMAs

    // fragments (16-bit A layout: VGPR v -> K pair at
    //            k = (v&3)*2 + (v>=4?16:0) + (lane>=16?8:0))
    v16h af[MI], bf[NJ];
#pragma unroll
    for (int i = 0; i < MI; ++i) {
      int mrow = wm * (MI * 16) + i * 16 + lx;
#pragma unroll
      for (int v = 0; v < 8; ++v) {
        int k = ((v & 3) << 1) + ((v >> 2) << 4) + (half << 3);
        af[i][2 * v]     = sA[cur * SA_SZ + mrow * 40 + k];
        af[i][2 * v + 1] = sA[cur * SA_SZ + mrow * 40 + k + 1];
      }
    }
#pragma unroll
    for (int j = 0; j < NJ; ++j) {
      int ncol = wn * (NJ * 16) + j * 16 + lx;
#pragma unroll
      for (int v = 0; v < 8; ++v) {
        int k = ((v & 3) << 1) + ((v >> 2) << 4) + (half << 3);
        if (BT) {
          bf[j][2 * v]     = sB[cur * SB_SZ + ncol * 40 + k];
          bf[j][2 * v + 1] = sB[cur * SB_SZ + ncol * 40 + k + 1];
        } else {
          bf[j][2 * v]     = sB[cur * SB_SZ + k * SB_STR + ncol];
          bf[j][2 * v + 1] = sB[cur * SB_SZ + (k + 1) * SB_STR + ncol];
        }
      }
    }
#pragma unroll
    for (int i = 0; i < MI; ++i)
#pragma unroll
      for (int j = 0; j < NJ; ++j)
        acc[i][j] = __builtin_amdgcn_wmma_f32_16x16x32_f16(
            false, af[i], false, bf[j], (short)0, acc[i][j], false, false);

    if (more) sstore(nxt);                 // nxt is not read by anyone this step
    __syncthreads();                       // single barrier per K-step
  }

  // ---- epilogue: C layout row = r + 8*half, col = lx ----
#pragma unroll
  for (int i = 0; i < MI; ++i)
#pragma unroll
    for (int j = 0; j < NJ; ++j) {
      int ncol = tn0 + wn * (NJ * 16) + j * 16 + lx;
      float bv = bias ? bias[ncol] : 0.0f;
#pragma unroll
      for (int r = 0; r < 8; ++r) {
        int mrow = tm0 + wm * (MI * 16) + i * 16 + r + (half << 3);
        float val = acc[i][j][r] * alpha + bv;
        if (act == 1) val = fmaxf(val, 0.0f);
        else if (act == 2) val = tanhf(val);
        if (Cf) Cf[(size_t)mrow * ldc + ncol] = val;
        if (Ch) Ch[(size_t)mrow * ldc + ncol] = (_Float16)val;
      }
    }
}

// ---------------------------- GEMM wrappers --------------------------------
// Main NT config: block 128x128, wave tile 64x32 (8 WMMA / wave / K-step).
__global__ void gemm_nt(const _Float16* __restrict__ A, const _Float16* __restrict__ W,
                        const float* __restrict__ bias, float* __restrict__ Cf,
                        _Float16* __restrict__ Ch,
                        int K, int lda, int ldw, int ldc, float alpha, int act) {
  gemm_core<1, 2, 4, 4, 2>(A, W, bias, Cf, Ch, K, lda, ldw, ldc, alpha, act,
                           blockIdx.y * 128, blockIdx.x * 128);
}

// scores[z, q, k] = (1/8) * q_z . k_z  over DH, z = b*H + h
__global__ void attn_qk(const _Float16* __restrict__ qkvh, float* __restrict__ scores) {
  int z = blockIdx.z, b = z / Hn, h = z % Hn;
  const _Float16* q = qkvh + (size_t)b * Ssz * 3 * Esz + h * DHn;
  const _Float16* k = q + Esz;
  float* c = scores + (size_t)z * Ssz * Ssz;
  gemm_core<1, 2, 4, 4, 2>(q, k, nullptr, c, nullptr, DHn, 3 * Esz, 3 * Esz, Ssz,
                           0.125f, 0, blockIdx.y * 128, blockIdx.x * 128);
}

// o[b, q, h*64+d] = sum_k P[z,q,k] * v[z,k,d]   (NN gemm, N = DH = 64)
__global__ void attn_pv(const _Float16* __restrict__ P, const _Float16* __restrict__ qkvh,
                        _Float16* __restrict__ O) {
  int z = blockIdx.z, b = z / Hn, h = z % Hn;
  const _Float16* p = P + (size_t)z * Ssz * Ssz;
  const _Float16* v = qkvh + (size_t)b * Ssz * 3 * Esz + 2 * Esz + h * DHn;
  _Float16* o = O + (size_t)b * Ssz * Esz + h * DHn;
  gemm_core<0, 4, 2, 2, 2>(p, v, nullptr, nullptr, o, Ssz, Ssz, 3 * Esz, Esz,
                           1.0f, 0, blockIdx.y * 128, blockIdx.x * 64);
}

// ---------------------------- elementwise kernels --------------------------
__global__ void f32_to_f16(const float* __restrict__ a, _Float16* __restrict__ o, long long n) {
  long long i = (long long)blockIdx.x * blockDim.x + threadIdx.x;
  long long stride = (long long)gridDim.x * blockDim.x;
  for (; i < n; i += stride) o[i] = (_Float16)a[i];
}

// x = LN(tok + pe[pos] + type_emb[type]); writes f32 + f16 mirrors.
__global__ void embed_ln(const float* __restrict__ tok, const int* __restrict__ pos,
                         const int* __restrict__ type, const float* __restrict__ pe,
                         const float* __restrict__ temb, const float* __restrict__ w,
                         const float* __restrict__ bvec,
                         float* __restrict__ x, _Float16* __restrict__ xh) {
  __shared__ float red[256];
  int row = blockIdx.x, i = threadIdx.x;
  int p = pos[row], ty = type[row];
  size_t base = (size_t)row * Esz;
  float v0 = tok[base + i]       + pe[(size_t)p * Esz + i]       + temb[(size_t)ty * Esz + i];
  float v1 = tok[base + 256 + i] + pe[(size_t)p * Esz + 256 + i] + temb[(size_t)ty * Esz + 256 + i];
  float v2 = tok[base + 512 + i] + pe[(size_t)p * Esz + 512 + i] + temb[(size_t)ty * Esz + 512 + i];
  float m  = block_sum(v0 + v1 + v2, red) * (1.0f / Esz);
  float d0 = v0 - m, d1 = v1 - m, d2 = v2 - m;
  float var = block_sum(d0 * d0 + d1 * d1 + d2 * d2, red) * (1.0f / Esz);
  float inv = rsqrtf(var + 1e-5f);
  float y0 = d0 * inv * w[i]       + bvec[i];
  float y1 = d1 * inv * w[256 + i] + bvec[256 + i];
  float y2 = d2 * inv * w[512 + i] + bvec[512 + i];
  x[base + i] = y0; x[base + 256 + i] = y1; x[base + 512 + i] = y2;
  xh[base + i] = (_Float16)y0; xh[base + 256 + i] = (_Float16)y1; xh[base + 512 + i] = (_Float16)y2;
}

// out = LN(xin [+ dd]); optionally updates xout (f32) and hout (f16).
__global__ void add_ln(const float* __restrict__ xin, const float* __restrict__ dd,
                       const float* __restrict__ w, const float* __restrict__ bvec,
                       float* __restrict__ xout, _Float16* __restrict__ hout) {
  __shared__ float red[256];
  int row = blockIdx.x, i = threadIdx.x;
  size_t base = (size_t)row * Esz;
  float v0 = xin[base + i],       v1 = xin[base + 256 + i],  v2 = xin[base + 512 + i];
  if (dd) { v0 += dd[base + i]; v1 += dd[base + 256 + i]; v2 += dd[base + 512 + i]; }
  float m  = block_sum(v0 + v1 + v2, red) * (1.0f / Esz);
  float d0 = v0 - m, d1 = v1 - m, d2 = v2 - m;
  float var = block_sum(d0 * d0 + d1 * d1 + d2 * d2, red) * (1.0f / Esz);
  float inv = rsqrtf(var + 1e-5f);
  float y0 = d0 * inv * w[i]       + bvec[i];
  float y1 = d1 * inv * w[256 + i] + bvec[256 + i];
  float y2 = d2 * inv * w[512 + i] + bvec[512 + i];
  if (xout) { xout[base + i] = y0; xout[base + 256 + i] = y1; xout[base + 512 + i] = y2; }
  if (hout) { hout[base + i] = (_Float16)y0; hout[base + 256 + i] = (_Float16)y1;
              hout[base + 512 + i] = (_Float16)y2; }
}

// row-wise masked softmax over key dim; one (b,h,q) row per block.
__global__ void attn_softmax(const float* __restrict__ scores,
                             const unsigned char* __restrict__ amask,
                             _Float16* __restrict__ P) {
  __shared__ float red[256];
  int row = blockIdx.x;               // z*S + q,  z = b*H + h
  int b = row / (Hn * Ssz);
  int i = threadIdx.x;
  float v = amask[(size_t)b * Ssz + i] ? NEGV : scores[(size_t)row * Ssz + i];
  float mx = block_max(v, red);
  float e = expf(v - mx);
  float s = block_sum(e, red);
  P[(size_t)row * Ssz + i] = (_Float16)(e / s);
}

// cos[b,s] = <tok, out> / max(|tok||out|, 1e-8)
__global__ void cos_kernel(const float* __restrict__ tok, const float* __restrict__ dv,
                           float* __restrict__ cosv) {
  __shared__ float red[256];
  int row = blockIdx.x, i = threadIdx.x;
  size_t base = (size_t)row * Esz;
  float dot = 0.f, na = 0.f, nb = 0.f;
#pragma unroll
  for (int c = 0; c < 3; ++c) {
    float a = tok[base + c * 256 + i], o = dv[base + c * 256 + i];
    dot += a * o; na += a * a; nb += o * o;
  }
  dot = block_sum(dot, red);
  na  = block_sum(na, red);
  nb  = block_sum(nb, red);
  if (i == 0) cosv[row] = dot / fmaxf(sqrtf(na) * sqrtf(nb), 1e-8f);
}

// masked softmaxes + gumbel logits + exact top-k ranking (forward of the
// straight-through estimator reduces to y_hard * valid). One batch per block.
__global__ void mask_gen(const float* __restrict__ cosv, const int* __restrict__ type,
                         const float* __restrict__ gh, const float* __restrict__ gt,
                         float* __restrict__ out) {
  __shared__ float red[256];
  __shared__ float glh[256];
  __shared__ float glt[256];
  int b = blockIdx.x, i = threadIdx.x;
  int ty = type[(size_t)b * Ssz + i];
  float c = cosv[(size_t)b * Ssz + i];
  bool v1 = (ty == 1), v2 = (ty == 2);
  float e1 = v1 ? expf(c) : 0.0f;
  float e2 = v2 ? expf(c) : 0.0f;
  float s1 = block_sum(e1, red);
  float s2 = block_sum(e2, red);
  float hp = e1 / s1;              // valid-only masked softmax
  float tp = 1.0f - e2 / s2;       // INFERENCE == False path
  glh[i] = v1 ? (logf(hp) + gh[(size_t)b * Ssz + i]) : NEGV;   // TAU == 1
  glt[i] = v2 ? (logf(tp) + gt[(size_t)b * Ssz + i]) : NEGV;
  __syncthreads();
  float gi1 = glh[i], gi2 = glt[i];
  int r1 = 0, r2 = 0;
  for (int j = 0; j < Ssz; ++j) {     // stable rank == jax.lax.top_k ordering
    float a = glh[j]; r1 += (a > gi1) || (a == gi1 && j < i);
    float t = glt[j]; r2 += (t > gi2) || (t == gi2 && j < i);
  }
  float o = ((v1 && r1 < KEEPH) ? 1.0f : 0.0f) + ((v2 && r2 < KEEPT) ? 1.0f : 0.0f);
  out[(size_t)b * Ssz + i] = o;
}

// ------------------------------- launcher ----------------------------------
extern "C" void kernel_launch(void* const* d_in, const int* in_sizes, int n_in,
                              void* d_out, int out_size, void* d_ws, size_t ws_size,
                              hipStream_t stream) {
  const float* tok   = (const float*)d_in[0];
  const unsigned char* amask = (const unsigned char*)d_in[1];
  const int* pos     = (const int*)d_in[2];
  const int* type    = (const int*)d_in[3];
  const float* gh    = (const float*)d_in[4];
  const float* gt    = (const float*)d_in[5];
  const float* pe    = (const float*)d_in[6];
  const float* temb  = (const float*)d_in[7];
  const float* ln_w  = (const float*)d_in[8];
  const float* ln_b  = (const float*)d_in[9];
  const float* dense_w = (const float*)d_in[10];
  const float* dense_b = (const float*)d_in[11];
  const float* qkv_w = (const float*)d_in[12];
  const float* qkv_b = (const float*)d_in[13];
  const float* out_w = (const float*)d_in[14];
  const float* out_b = (const float*)d_in[15];
  const float* ln1_w = (const float*)d_in[16];
  const float* ln1_b = (const float*)d_in[17];
  const float* lin1_w = (const float*)d_in[18];
  const float* lin1_b = (const float*)d_in[19];
  const float* lin2_w = (const float*)d_in[20];
  const float* lin2_b = (const float*)d_in[21];
  const float* ln2_w = (const float*)d_in[22];
  const float* ln2_b = (const float*)d_in[23];

  char* ws = (char*)d_ws;
  size_t off = 0;
  auto alloc = [&](size_t bytes) -> char* {
    off = (off + 255) & ~(size_t)255;
    char* p = ws + off;
    off += bytes;
    return p;
  };

  float*    x      = (float*)   alloc((size_t)ROWS * Esz * 4);
  _Float16* xh     = (_Float16*)alloc((size_t)ROWS * Esz * 2);
  _Float16* qkvh   = (_Float16*)alloc((size_t)ROWS * 3 * Esz * 2);
  float*    scores = (float*)   alloc((size_t)Bsz * Hn * Ssz * Ssz * 4); // reused as f1h
  _Float16* Ph     = (_Float16*)alloc((size_t)Bsz * Hn * Ssz * Ssz * 2);
  _Float16* oh     = (_Float16*)alloc((size_t)ROWS * Esz * 2);           // reused as outln_h
  float*    tmp    = (float*)   alloc((size_t)ROWS * Esz * 4);           // reused as dvec
  float*    cosb   = (float*)   alloc((size_t)ROWS * 4);
  _Float16* wqkv   = (_Float16*)alloc((size_t)Lnum * 3 * Esz * Esz * 2);
  _Float16* wout   = (_Float16*)alloc((size_t)Lnum * Esz * Esz * 2);
  _Float16* wlin1  = (_Float16*)alloc((size_t)Lnum * FFn * Esz * 2);
  _Float16* wlin2  = (_Float16*)alloc((size_t)Lnum * Esz * FFn * 2);
  _Float16* wdense = (_Float16*)alloc((size_t)Esz * Esz * 2);
  _Float16* f1h    = (_Float16*)scores;   // FFN intermediate overlays dead scores
  _Float16* outln  = oh;
  float*    dvec   = tmp;

  // weight f32 -> f16
  f32_to_f16<<<1024, 256, 0, stream>>>(qkv_w,  wqkv,  (long long)Lnum * 3 * Esz * Esz);
  f32_to_f16<<<1024, 256, 0, stream>>>(out_w,  wout,  (long long)Lnum * Esz * Esz);
  f32_to_f16<<<1024, 256, 0, stream>>>(lin1_w, wlin1, (long long)Lnum * FFn * Esz);
  f32_to_f16<<<1024, 256, 0, stream>>>(lin2_w, wlin2, (long long)Lnum * Esz * FFn);
  f32_to_f16<<<1024, 256, 0, stream>>>(dense_w, wdense, (long long)Esz * Esz);

  embed_ln<<<ROWS, 256, 0, stream>>>(tok, pos, type, pe, temb, ln_w, ln_b, x, xh);

  for (int l = 0; l < Lnum; ++l) {
    // QKV projection: [8192,768] @ [2304,768]^T -> f16 qkv
    gemm_nt<<<dim3(3 * Esz / 128, ROWS / 128, 1), 256, 0, stream>>>(
        xh, wqkv + (size_t)l * 3 * Esz * Esz, qkv_b + (size_t)l * 3 * Esz,
        nullptr, qkvh, Esz, Esz, Esz, 3 * Esz, 1.0f, 0);
    // scores = q k^T / sqrt(DH), batched over 384 (b,h)
    attn_qk<<<dim3(Ssz / 128, Ssz / 128, Bsz * Hn), 256, 0, stream>>>(qkvh, scores);
    attn_softmax<<<Bsz * Hn * Ssz, 256, 0, stream>>>(scores, amask, Ph);
    // o = P v
    attn_pv<<<dim3(DHn / 64, Ssz / 128, Bsz * Hn), 256, 0, stream>>>(Ph, qkvh, oh);
    // out projection -> tmp (f32), then residual + LN1
    gemm_nt<<<dim3(Esz / 128, ROWS / 128, 1), 256, 0, stream>>>(
        oh, wout + (size_t)l * Esz * Esz, out_b + (size_t)l * Esz,
        tmp, nullptr, Esz, Esz, Esz, Esz, 1.0f, 0);
    add_ln<<<ROWS, 256, 0, stream>>>(x, tmp, ln1_w + (size_t)l * Esz,
                                     ln1_b + (size_t)l * Esz, x, xh);
    // FFN: relu(x W1^T + b1) W2^T + b2, then residual + LN2
    gemm_nt<<<dim3(FFn / 128, ROWS / 128, 1), 256, 0, stream>>>(
        xh, wlin1 + (size_t)l * FFn * Esz, lin1_b + (size_t)l * FFn,
        nullptr, f1h, Esz, Esz, Esz, FFn, 1.0f, 1);
    gemm_nt<<<dim3(Esz / 128, ROWS / 128, 1), 256, 0, stream>>>(
        f1h, wlin2 + (size_t)l * Esz * FFn, lin2_b + (size_t)l * Esz,
        tmp, nullptr, FFn, FFn, FFn, Esz, 1.0f, 0);
    add_ln<<<ROWS, 256, 0, stream>>>(x, tmp, ln2_w + (size_t)l * Esz,
                                     ln2_b + (size_t)l * Esz, x, xh);
  }

  // final LN -> f16, dense + tanh -> f32, cosine, mask head
  add_ln<<<ROWS, 256, 0, stream>>>(x, nullptr, ln_w, ln_b, nullptr, outln);
  gemm_nt<<<dim3(Esz / 128, ROWS / 128, 1), 256, 0, stream>>>(
      outln, wdense, dense_b, dvec, nullptr, Esz, Esz, Esz, Esz, 1.0f, 2);
  cos_kernel<<<ROWS, 256, 0, stream>>>(tok, dvec, cosb);
  mask_gen<<<Bsz, 256, 0, stream>>>(cosb, type, gh, gt, (float*)d_out);
}
